// CausalSelfAttention_68461778698773
// MI455X (gfx1250) — matrix operand order repaired and checked by reading, not compiled, not run on hardware
//
#include <hip/hip_runtime.h>
#include <hip/hip_bf16.h>

// ---------------------------------------------------------------------------
// CausalSelfAttention (quirky variant) for MI455X / gfx1250, bf16 WMMA path.
//   B=4, S=2048, D=1024
//   q,k,v = x @ W{q,k,v}.T
//   st[n,m] = (k @ q.T)/sqrt(D)          (scores, pre-transposed)
//   a_t     = row softmax of st (== softmax over query axis), mask m<=n
//   out[n,:] = sum_m a_t[n,m] * v[m,:]
// All three GEMM stages are "NT" (both operands K-contiguous) so both A and B
// WMMA fragments are filled with plain contiguous global_load_b128s.
// ---------------------------------------------------------------------------

#define BATCH 4
#define SEQ   2048
#define DIM   1024

typedef __attribute__((ext_vector_type(16))) __bf16        v16bf;
typedef __attribute__((ext_vector_type(8)))  float         v8f;
typedef __attribute__((ext_vector_type(4)))  unsigned int  v4u;   // 128-bit

// Global (addrspace(1)) pointer types: force global_load/global_store
// selection instead of flat_load/flat_store, so fragment loads only tick
// LOADcnt (not DScnt).
typedef const __attribute__((address_space(1))) v4u*           gcp_b128;
typedef __attribute__((address_space(1))) float*               gp_f32;
typedef __attribute__((address_space(1))) unsigned short*      gp_u16;

union Frag {
    v16bf v;
    v4u   q[2];
};

__device__ __forceinline__ unsigned short f2bf(float f) {
    unsigned int u = __float_as_uint(f);
    u += 0x7fffu + ((u >> 16) & 1u);   // round-to-nearest-even
    return (unsigned short)(u >> 16);
}

// -------------------------------- f32 -> bf16 ------------------------------
__global__ void cvt_f32_bf16_kernel(const float* __restrict__ in,
                                    unsigned short* __restrict__ out, int n) {
    int i = blockIdx.x * blockDim.x + threadIdx.x;
    if (i < n) out[i] = f2bf(in[i]);
}

// ------------------------- bf16 NT GEMM, WMMA core -------------------------
// C[i,j] = alpha * sum_t A[i,t] * B[j,t]
// Block = 128 threads (4 waves), block tile 128x128, wave tile 64x64.
// Per K-step: 16 global_load_b128 (8 frags) then 16 back-to-back WMMAs, so
// the scheduler can overlap each step's loads with the previous step's WMMA
// block. launch_bounds(.,1): full VGPR allocation, no accumulator spills.
template <bool BF16_OUT>
__global__ __launch_bounds__(128, 1)
void gemm_nt_kernel(const unsigned short* __restrict__ A,
                    const unsigned short* __restrict__ Bm,
                    void* __restrict__ Cv,
                    int K, int lda, int ldb, int ldc, float alpha,
                    long long sA, long long sB, long long sC) {
    const int b = blockIdx.z;
    A  += (size_t)b * (size_t)sA;
    Bm += (size_t)b * (size_t)sB;
    float*          Cf = (float*)Cv          + (size_t)b * (size_t)sC;
    unsigned short* Cb = (unsigned short*)Cv + (size_t)b * (size_t)sC;

    const int tid  = threadIdx.x;
    const int wave = tid >> 5;
    const int lane = tid & 31;
    const int lr   = lane & 15;   // row/col within 16-tile
    const int lh   = lane >> 4;   // half select

    const int wm = blockIdx.y * 128 + (wave >> 1) * 64;  // wave M origin
    const int wn = blockIdx.x * 128 + (wave & 1) * 64;   // wave N origin

    // Hoisted per-row base pointers; bumped by 32 elements (64 B) per K step.
    const unsigned short* pa[4];
    const unsigned short* pb[4];
#pragma unroll
    for (int i = 0; i < 4; ++i)
        pa[i] = A + (size_t)(wm + i * 16 + lr) * lda + lh * 8;
#pragma unroll
    for (int j = 0; j < 4; ++j)
        pb[j] = Bm + (size_t)(wn + j * 16 + lr) * ldb + lh * 8;

    v8f acc[4][4] = {};

    for (int k0 = 0; k0 < K; k0 += 32) {
        // All 8 fragments for this K-step loaded up front (16 b128 loads),
        // then 16 WMMAs with no loads interleaved.
        Frag bf[4];
        Frag af[4];
#pragma unroll
        for (int j = 0; j < 4; ++j) {
            // B fragment (32x16): lane n holds K = lh*8..+7 (elems 0-7) and
            // K = 16+lh*8..+7 (elems 8-15).
            bf[j].q[0] = *(gcp_b128)(pb[j]);
            bf[j].q[1] = *(gcp_b128)(pb[j] + 16);
        }
#pragma unroll
        for (int i = 0; i < 4; ++i) {
            // A fragment (16x32): lane m holds K = lh*8..+7 (elems 0-7) and
            // K = 16+lh*8..+7 (elems 8-15).
            af[i].q[0] = *(gcp_b128)(pa[i]);
            af[i].q[1] = *(gcp_b128)(pa[i] + 16);
        }
#pragma unroll
        for (int i = 0; i < 4; ++i) {
#pragma unroll
            for (int j = 0; j < 4; ++j) {
                acc[i][j] = __builtin_amdgcn_wmma_f32_16x16x32_bf16(
                    false, af[i].v, false, bf[j].v, (short)0, acc[i][j],
                    false, false);
            }
        }
#pragma unroll
        for (int i = 0; i < 4; ++i) pa[i] += 32;
#pragma unroll
        for (int j = 0; j < 4; ++j) pb[j] += 32;
    }

    // C/D 16x16 f32 layout: VGPR r -> row = r + (lane/16)*8, col = lane%16
#pragma unroll
    for (int i = 0; i < 4; ++i) {
#pragma unroll
        for (int j = 0; j < 4; ++j) {
            const int col   = wn + j * 16 + lr;
            const int rbase = wm + i * 16 + lh * 8;
#pragma unroll
            for (int r = 0; r < 8; ++r) {
                const float  val = acc[i][j][r] * alpha;
                const size_t off = (size_t)(rbase + r) * ldc + col;
                if (BF16_OUT) *(gp_u16)(Cb + off) = f2bf(val);
                else          *(gp_f32)(Cf + off) = val;
            }
        }
    }
}

// ------------- row softmax of st (axis=-2 of original scores) --------------
// block = one row n of st[b]; softmax over full row, then zero m > n (triu).
__global__ void softmax_mask_kernel(const float* __restrict__ st,
                                    unsigned short* __restrict__ at, int S) {
    const int row = blockIdx.x;     // b*S + n
    const int n   = row % S;
    const float* r          = st + (size_t)row * S;
    unsigned short* o       = at + (size_t)row * S;
    __shared__ float red[256];
    const int tid = threadIdx.x;

    float mx = -3.402823466e38f;
    for (int m = tid; m < S; m += 256) mx = fmaxf(mx, r[m]);
    red[tid] = mx;
    __syncthreads();
    for (int s = 128; s > 0; s >>= 1) {
        if (tid < s) red[tid] = fmaxf(red[tid], red[tid + s]);
        __syncthreads();
    }
    mx = red[0];
    __syncthreads();

    float sum = 0.f;
    for (int m = tid; m < S; m += 256) sum += __expf(r[m] - mx);
    red[tid] = sum;
    __syncthreads();
    for (int s = 128; s > 0; s >>= 1) {
        if (tid < s) red[tid] += red[tid + s];
        __syncthreads();
    }
    const float inv = 1.0f / red[0];

    for (int m = tid; m < S; m += 256) {
        const float a = __expf(r[m] - mx) * inv;
        o[m] = (m <= n) ? f2bf(a) : (unsigned short)0;
    }
}

// ------------------------- bf16 transpose (v -> vt) ------------------------
// in: [b, R, C] -> out: [b, C, R]; LDS 32x32 tile, block (32,8).
__global__ void transpose_bf16_kernel(const unsigned short* __restrict__ in,
                                      unsigned short* __restrict__ out,
                                      int R, int C) {
    __shared__ unsigned short tile[32][33];
    const int b  = blockIdx.z;
    const int r0 = blockIdx.y * 32;
    const int c0 = blockIdx.x * 32;
    const unsigned short* ib = in  + (size_t)b * R * C;
    unsigned short*       ob = out + (size_t)b * R * C;

    for (int i = threadIdx.y; i < 32; i += 8)
        tile[i][threadIdx.x] = ib[(size_t)(r0 + i) * C + c0 + threadIdx.x];
    __syncthreads();
    for (int i = threadIdx.y; i < 32; i += 8)
        ob[(size_t)(c0 + i) * R + r0 + threadIdx.x] = tile[threadIdx.x][i];
}

// ---------------------------------------------------------------------------
extern "C" void kernel_launch(void* const* d_in, const int* in_sizes, int n_in,
                              void* d_out, int out_size, void* d_ws,
                              size_t ws_size, hipStream_t stream) {
    const float* x  = (const float*)d_in[0];   // [B,S,D]
    const float* wq = (const float*)d_in[1];   // [D,D]
    const float* wk = (const float*)d_in[2];
    const float* wv = (const float*)d_in[3];
    float* out = (float*)d_out;                // [B,S,D]

    const int nX = BATCH * SEQ * DIM;          // 8,388,608
    const int nW = DIM * DIM;                  // 1,048,576

    // Workspace layout (bytes) — total ~182 MB
    char* ws = (char*)d_ws;
    unsigned short* xb  = (unsigned short*)(ws);                       // 16 MB
    unsigned short* wqb = (unsigned short*)(ws + 16777216);            //  2 MB
    unsigned short* wkb = (unsigned short*)(ws + 16777216 + 2097152);
    unsigned short* wvb = (unsigned short*)(ws + 16777216 + 2 * 2097152);
    unsigned short* qb  = (unsigned short*)(ws + 16777216 + 3 * 2097152);
    unsigned short* kb  = qb + (size_t)nX;
    unsigned short* vb  = kb + (size_t)nX;
    unsigned short* vt  = vb + (size_t)nX;
    float*          st  = (float*)(vt + (size_t)nX);                   // 64 MB
    unsigned short* at  = (unsigned short*)(st + (size_t)BATCH * SEQ * SEQ);

    // 1) f32 -> bf16 conversions
    cvt_f32_bf16_kernel<<<(nX + 255) / 256, 256, 0, stream>>>(x, xb, nX);
    cvt_f32_bf16_kernel<<<(nW + 255) / 256, 256, 0, stream>>>(wq, wqb, nW);
    cvt_f32_bf16_kernel<<<(nW + 255) / 256, 256, 0, stream>>>(wk, wkb, nW);
    cvt_f32_bf16_kernel<<<(nW + 255) / 256, 256, 0, stream>>>(wv, wvb, nW);

    // 2) q/k/v = x @ W.T   (M=B*S=8192, N=D, K=D), bf16 out
    {
        dim3 grid(DIM / 128, (BATCH * SEQ) / 128, 1);
        gemm_nt_kernel<true><<<grid, 128, 0, stream>>>(
            xb, wqb, (void*)qb, DIM, DIM, DIM, DIM, 1.0f, 0, 0, 0);
        gemm_nt_kernel<true><<<grid, 128, 0, stream>>>(
            xb, wkb, (void*)kb, DIM, DIM, DIM, DIM, 1.0f, 0, 0, 0);
        gemm_nt_kernel<true><<<grid, 128, 0, stream>>>(
            xb, wvb, (void*)vb, DIM, DIM, DIM, DIM, 1.0f, 0, 0, 0);
    }

    // 3) st[b,n,m] = (k_b @ q_b.T)/sqrt(D)  (M=N=S, K=D), f32 out
    {
        dim3 grid(SEQ / 128, SEQ / 128, BATCH);
        gemm_nt_kernel<false><<<grid, 128, 0, stream>>>(
            kb, qb, (void*)st, DIM, DIM, DIM, SEQ, 0.03125f,
            (long long)SEQ * DIM, (long long)SEQ * DIM, (long long)SEQ * SEQ);
    }

    // 4) row softmax of st + triu mask -> a_t bf16
    softmax_mask_kernel<<<BATCH * SEQ, 256, 0, stream>>>(st, at, SEQ);

    // 5) vt[b,d,m] = v[b,m,d]
    {
        dim3 grid(DIM / 32, SEQ / 32, BATCH);
        transpose_bf16_kernel<<<grid, dim3(32, 8), 0, stream>>>(vb, vt, SEQ, DIM);
    }

    // 6) out[b,n,d] = sum_m a_t[n,m] * vt[d,m]  (M=S, N=D, K=S), f32 -> d_out
    {
        dim3 grid(DIM / 128, SEQ / 128, BATCH);
        gemm_nt_kernel<false><<<grid, 128, 0, stream>>>(
            at, vt, (void*)out, SEQ, SEQ, SEQ, DIM, 1.0f,
            (long long)SEQ * SEQ, (long long)DIM * SEQ, (long long)SEQ * DIM);
    }
}
